// MultiHeadAttention_48146583388304
// MI455X (gfx1250) — matrix operand order repaired and checked by reading, not compiled, not run on hardware
//
#include <hip/hip_runtime.h>

// ---------- CDNA5 WMMA types ----------
typedef __attribute__((ext_vector_type(16))) _Float16 v16h;
typedef __attribute__((ext_vector_type(8)))  float    v8f;

#define N_TOK  4096   // B*S
#define DMODEL 1024
#define SEQ    2048
#define DPH    64
#define NHMAT  32     // B*H
#define LN_EPS 1e-5f
// softmax scale 0.5 is folded into the Q operand (exact in f16)

union U16 { v16h v; _Float16 e[16]; uint4 q[2]; };

static __device__ __forceinline__ v8f wmma16(v16h a, v16h b, v8f c) {
  // D = A(16x32 f16) x B(32x16 f16) + C(16x16 f32)
  return __builtin_amdgcn_wmma_f32_16x16x32_f16(false, a, false, b, (short)0, c, false, false);
}

template <typename T> struct is_f32 { static constexpr bool value = false; };
template <> struct is_f32<float>   { static constexpr bool value = true;  };

// ---------- 1) Weight prep: Wt[k][o] = (f16) W[o][k], 4 matrices ----------
__global__ __launch_bounds__(256) void prep_weights(
    const float* __restrict__ Wq, const float* __restrict__ Wk,
    const float* __restrict__ Wv, const float* __restrict__ Wo,
    _Float16* __restrict__ Wt) {
  const float* W[4] = {Wq, Wk, Wv, Wo};
  const int w   = blockIdx.y;
  const int idx = blockIdx.x * 256 + threadIdx.x;   // 0..1M-1
  const int k = idx & (DMODEL - 1);
  const int o = idx >> 10;
  Wt[(size_t)w * DMODEL * DMODEL + (size_t)k * DMODEL + o] =
      (_Float16)W[w][(size_t)o * DMODEL + k];
}

// ---------- 2) GEMM: C[n][o] = X[n][:] . Wt[:][o] + bias[o] ----------
// MODE 0: store f16 row-major   (Q, V projections)
// MODE 1: store f16 per-head-transposed Kt[bh][d][s]  (K projection)
// MODE 2: store f32 row-major   (output projection, pre-LN)
template <typename AT, int MODE>
__global__ __launch_bounds__(256) void proj_gemm(
    const AT* __restrict__ X, const _Float16* __restrict__ Wt,
    const float* __restrict__ bias, void* __restrict__ outp) {
  const int lane = threadIdx.x & 31;
  const int warp = threadIdx.x >> 5;
  const int tile = blockIdx.x * 8 + warp;       // 0..4095
  const int m0 = (tile >> 4) * 16;              // 256 row tiles
  const int o0 = (tile & 15) * 64;              // 16 col groups of 64
  const int col = lane & 15;
  const int hl  = lane >> 4;
  const int abase = hl ? 8 : 0;

  v8f acc[4] = {};
  const AT* arow = X + (size_t)(m0 + col) * DMODEL;

  for (int kk = 0; kk < DMODEL; kk += 32) {
    U16 A;
    if constexpr (is_f32<AT>::value) {
      const float* p0 = arow + kk + abase;
#pragma unroll
      for (int j = 0; j < 8; ++j) A.e[j]     = (_Float16)p0[j];
#pragma unroll
      for (int j = 0; j < 8; ++j) A.e[8 + j] = (_Float16)p0[16 + j];
    } else {
      A.q[0] = *(const uint4*)(arow + kk + abase);
      A.q[1] = *(const uint4*)(arow + kk + abase + 16);
    }
#pragma unroll
    for (int g = 0; g < 4; ++g) {
      U16 B;
      const _Float16* bp = Wt + (size_t)(kk + abase) * DMODEL + o0 + g * 16 + col;
#pragma unroll
      for (int j = 0; j < 8; ++j) {
        B.e[j]     = bp[(size_t)j * DMODEL];
        B.e[8 + j] = bp[(size_t)(16 + j) * DMODEL];
      }
      acc[g] = wmma16(A.v, B.v, acc[g]);
    }
  }

  float bval[4];
#pragma unroll
  for (int g = 0; g < 4; ++g) bval[g] = bias[o0 + g * 16 + col];

#pragma unroll
  for (int g = 0; g < 4; ++g) {
#pragma unroll
    for (int r = 0; r < 8; ++r) {
      const int n = m0 + r + 8 * hl;
      const int o = o0 + g * 16 + col;
      const float val = acc[g][r] + bval[g];
      if constexpr (MODE == 0) {
        ((_Float16*)outp)[(size_t)n * DMODEL + o] = (_Float16)val;
      } else if constexpr (MODE == 1) {
        // Kt[bh][d][s]: bh=n>>7, d=o&63, s=((n&127)<<4)|(o>>6)
        const size_t addr = ((size_t)(n >> 7) << 17) + ((size_t)(o & 63) << 11) +
                            ((size_t)(n & 127) << 4) + (size_t)(o >> 6);
        ((_Float16*)outp)[addr] = (_Float16)val;
      } else {
        ((float*)outp)[(size_t)n * DMODEL + o] = val;
      }
    }
  }
}

// ---------- 3) Fused attention: scores + softmax + attention write + P.V ----------
// One wave handles 32 q-rows: every K/V B-tile load feeds TWO WMMAs (halves L2 traffic).
__global__ __launch_bounds__(256) void attention_kernel(
    const _Float16* __restrict__ Qb, const _Float16* __restrict__ Kt,
    const _Float16* __restrict__ Vb, _Float16* __restrict__ Cb,
    float* __restrict__ attn) {
  __shared__ __align__(16) float pbuf[8][32 * 32];   // 4 KB per wave
  const int lane = threadIdx.x & 31;
  const int warp = threadIdx.x >> 5;
  const int wid  = blockIdx.x * 8 + warp;            // 0..2047
  const int bh = wid >> 6;                           // 32 heads
  const int q0 = (wid & 63) * 32;                    // 64 q-blocks of 32 rows
  const int col = lane & 15;
  const int hl  = lane >> 4;
  float* P = pbuf[warp];

  const _Float16* Qh = Qb + (size_t)bh * SEQ * DPH;  // [2048][64]
  const _Float16* Kh = Kt + (size_t)bh * SEQ * DPH;  // [64][2048] (transposed)
  const _Float16* Vh = Vb + (size_t)bh * SEQ * DPH;  // [2048][64]
  float* attn_h = attn + (size_t)bh * SEQ * SEQ;

  // Q A-tiles for two 16-row groups, K=64 split into two 16x32 operands.
  // Fold softmax SCALE=0.5 into Q (exact in f16).
  U16 aq[2][2];
#pragma unroll
  for (int rg = 0; rg < 2; ++rg) {
    const _Float16* qrow = Qh + (size_t)(q0 + rg * 16 + col) * DPH;
    const int b = hl ? 8 : 0;
    aq[rg][0].q[0] = *(const uint4*)(qrow + b);
    aq[rg][0].q[1] = *(const uint4*)(qrow + b + 16);
    aq[rg][1].q[0] = *(const uint4*)(qrow + 32 + b);
    aq[rg][1].q[1] = *(const uint4*)(qrow + 32 + b + 16);
#pragma unroll
    for (int h = 0; h < 2; ++h)
#pragma unroll
      for (int j = 0; j < 16; ++j)
        aq[rg][h].e[j] = (_Float16)((float)aq[rg][h].e[j] * 0.5f);
  }

  // Per-lane online softmax stats (this lane's own columns only).
  float mrow[2][8], lrow[2][8];
#pragma unroll
  for (int rg = 0; rg < 2; ++rg)
#pragma unroll
    for (int r = 0; r < 8; ++r) { mrow[rg][r] = -1e30f; lrow[rg][r] = 0.f; }

  // ---- pass 1: per-lane online row max / compensated sum (stats only) ----
  for (int kr = 0; kr < SEQ; kr += 32) {
    v8f st[2][2];                                    // [rg][t]
#pragma unroll
    for (int t = 0; t < 2; ++t) {
      U16 B0, B1;
      {
        const _Float16* bp0 = Kh + (size_t)(8 * hl) * SEQ + kr + t * 16 + col;
        const _Float16* bp1 = bp0 + (size_t)32 * SEQ;
        __builtin_prefetch(bp0 + 32, 0, 1);
        __builtin_prefetch(bp1 + 32, 0, 1);
#pragma unroll
        for (int j = 0; j < 8; ++j) {
          B0.e[j] = bp0[(size_t)j * SEQ];  B0.e[8 + j] = bp0[(size_t)(16 + j) * SEQ];
          B1.e[j] = bp1[(size_t)j * SEQ];  B1.e[8 + j] = bp1[(size_t)(16 + j) * SEQ];
        }
      }
#pragma unroll
      for (int rg = 0; rg < 2; ++rg) {
        v8f s = {};
        s = wmma16(aq[rg][0].v, B0.v, s);
        s = wmma16(aq[rg][1].v, B1.v, s);
        st[rg][t] = s;
      }
    }
#pragma unroll
    for (int rg = 0; rg < 2; ++rg)
#pragma unroll
      for (int r = 0; r < 8; ++r) {
        const float v0 = st[rg][0][r];
        const float v1 = st[rg][1][r];
        const float cm = fmaxf(v0, v1);
        const float mn = fmaxf(mrow[rg][r], cm);
        lrow[rg][r] = lrow[rg][r] * __expf(mrow[rg][r] - mn) +
                      __expf(v0 - mn) + __expf(v1 - mn);
        mrow[rg][r] = mn;
      }
  }

  // ---- single cross-lane compensated merge (4 xor steps over 16-lane group) ----
  float inv[2][8];
#pragma unroll
  for (int rg = 0; rg < 2; ++rg)
#pragma unroll
    for (int r = 0; r < 8; ++r) {
      float m = mrow[rg][r], l = lrow[rg][r];
#pragma unroll
      for (int mask = 1; mask <= 8; mask <<= 1) {
        const float om = __shfl_xor(m, mask, 32);
        const float ol = __shfl_xor(l, mask, 32);
        const float mn = fmaxf(m, om);
        l = l * __expf(m - mn) + ol * __expf(om - mn);
        m = mn;
      }
      mrow[rg][r] = m;
      inv[rg][r]  = 1.f / l;
    }

  // ---- pass 2: normalized attention out + context accumulate ----
  v8f ctx[2][4] = {};
  for (int kr = 0; kr < SEQ; kr += 32) {
#pragma unroll
    for (int t = 0; t < 2; ++t) {
      U16 B0, B1;
      {
        const _Float16* bp0 = Kh + (size_t)(8 * hl) * SEQ + kr + t * 16 + col;
        const _Float16* bp1 = bp0 + (size_t)32 * SEQ;
        __builtin_prefetch(bp0 + 32, 0, 1);
        __builtin_prefetch(bp1 + 32, 0, 1);
#pragma unroll
        for (int j = 0; j < 8; ++j) {
          B0.e[j] = bp0[(size_t)j * SEQ];  B0.e[8 + j] = bp0[(size_t)(16 + j) * SEQ];
          B1.e[j] = bp1[(size_t)j * SEQ];  B1.e[8 + j] = bp1[(size_t)(16 + j) * SEQ];
        }
      }
#pragma unroll
      for (int rg = 0; rg < 2; ++rg) {
        v8f s = {};
        s = wmma16(aq[rg][0].v, B0.v, s);
        s = wmma16(aq[rg][1].v, B1.v, s);
#pragma unroll
        for (int r = 0; r < 8; ++r) {
          const float p = __expf(s[r] - mrow[rg][r]) * inv[rg][r];
          P[(rg * 16 + r + 8 * hl) * 32 + t * 16 + col] = p;   // stage in LDS
        }
      }
    }
    asm volatile("s_wait_dscnt 0" ::: "memory");   // LDS stores visible to whole wave

    // coalesced attention write: lane -> (row = col, 16-col half = hl) per row group
#pragma unroll
    for (int rg = 0; rg < 2; ++rg) {
      const float* src = P + (rg * 16 + col) * 32 + hl * 16;
      float* dst = attn_h + (size_t)(q0 + rg * 16 + col) * SEQ + kr + hl * 16;
#pragma unroll
      for (int j = 0; j < 4; ++j)
        ((float4*)dst)[j] = ((const float4*)src)[j];
    }

    // re-read P in WMMA A layout (two 16x32 f16 tiles)
    U16 pa[2];
#pragma unroll
    for (int rg = 0; rg < 2; ++rg) {
      const float* pr = P + (rg * 16 + col) * 32;
      const int b = hl ? 8 : 0;
#pragma unroll
      for (int j = 0; j < 8; ++j) pa[rg].e[j]     = (_Float16)pr[b + j];
#pragma unroll
      for (int j = 0; j < 8; ++j) pa[rg].e[8 + j] = (_Float16)pr[b + 16 + j];
    }

    // ctx += P(32x32) x V(32x64): each V B-tile feeds two WMMAs
#pragma unroll
    for (int g = 0; g < 4; ++g) {
      U16 B;
      const _Float16* vp = Vh + (size_t)(kr + 8 * hl) * DPH + g * 16 + col;
#pragma unroll
      for (int j = 0; j < 8; ++j) {
        B.e[j]     = vp[(size_t)j * DPH];
        B.e[8 + j] = vp[(size_t)(16 + j) * DPH];
      }
#pragma unroll
      for (int rg = 0; rg < 2; ++rg)
        ctx[rg][g] = wmma16(pa[rg].v, B.v, ctx[rg][g]);
    }
  }

  // context -> Cb (contiguous per-head slab == [B,S,D] layout)
  _Float16* Ch = Cb + (size_t)bh * SEQ * DPH;
#pragma unroll
  for (int rg = 0; rg < 2; ++rg)
#pragma unroll
    for (int g = 0; g < 4; ++g)
#pragma unroll
      for (int r = 0; r < 8; ++r)
        Ch[(size_t)(q0 + rg * 16 + r + 8 * hl) * DPH + g * 16 + col] =
            (_Float16)ctx[rg][g][r];
}

// ---------- 4) residual + LayerNorm ----------
__global__ __launch_bounds__(256) void ln_kernel(
    const float* __restrict__ residual, const float* __restrict__ proj,
    const float* __restrict__ gamma, const float* __restrict__ beta,
    float* __restrict__ out) {
  __shared__ float red[256];
  const int row = blockIdx.x;
  const int tid = threadIdx.x;
  const size_t base = (size_t)row * DMODEL + tid * 4;

  float x[4];
#pragma unroll
  for (int i = 0; i < 4; ++i) x[i] = residual[base + i] + proj[base + i];

  float s = x[0] + x[1] + x[2] + x[3];
  red[tid] = s;
  __syncthreads();
  for (int off = 128; off; off >>= 1) {
    if (tid < off) red[tid] += red[tid + off];
    __syncthreads();
  }
  const float mu = red[0] * (1.f / DMODEL);
  __syncthreads();

  float sq = 0.f;
#pragma unroll
  for (int i = 0; i < 4; ++i) { const float d = x[i] - mu; sq += d * d; }
  red[tid] = sq;
  __syncthreads();
  for (int off = 128; off; off >>= 1) {
    if (tid < off) red[tid] += red[tid + off];
    __syncthreads();
  }
  const float rstd = rsqrtf(red[0] * (1.f / DMODEL) + LN_EPS);

#pragma unroll
  for (int i = 0; i < 4; ++i) {
    const int c = tid * 4 + i;
    out[base + i] = (x[i] - mu) * rstd * gamma[c] + beta[c];
  }
}

// ---------- launch ----------
extern "C" void kernel_launch(void* const* d_in, const int* in_sizes, int n_in,
                              void* d_out, int out_size, void* d_ws, size_t ws_size,
                              hipStream_t stream) {
  (void)in_sizes; (void)n_in; (void)out_size; (void)ws_size;
  const float* query = (const float*)d_in[0];
  const float* key   = (const float*)d_in[1];
  const float* value = (const float*)d_in[2];
  const float* Wq    = (const float*)d_in[3];
  const float* bq    = (const float*)d_in[4];
  const float* Wk    = (const float*)d_in[5];
  const float* bk    = (const float*)d_in[6];
  const float* Wv    = (const float*)d_in[7];
  const float* bv    = (const float*)d_in[8];
  const float* Wo    = (const float*)d_in[9];
  const float* bo    = (const float*)d_in[10];
  const float* gamma = (const float*)d_in[11];
  const float* beta  = (const float*)d_in[12];

  char* ws = (char*)d_ws;
  const size_t MB8 = 8u * 1024u * 1024u;
  _Float16* Qb  = (_Float16*)(ws + 0 * MB8);
  _Float16* Kt  = (_Float16*)(ws + 1 * MB8);
  _Float16* Vb  = (_Float16*)(ws + 2 * MB8);
  _Float16* Cb  = (_Float16*)(ws + 3 * MB8);
  _Float16* Wt  = (_Float16*)(ws + 4 * MB8);            // 4 x 2MB
  float*    prj = (float*)   (ws + 5 * MB8);            // 16MB

  float* out_ln   = (float*)d_out;                      // [2,2048,1024]
  float* out_attn = (float*)d_out + (size_t)2 * SEQ * DMODEL; // [32,2048,2048]

  prep_weights<<<dim3(4096, 4), 256, 0, stream>>>(Wq, Wk, Wv, Wo, Wt);
  proj_gemm<float, 0><<<512, 256, 0, stream>>>(query, Wt,               bq, (void*)Qb);
  proj_gemm<float, 1><<<512, 256, 0, stream>>>(key,   Wt + 1 * 1048576, bk, (void*)Kt);
  proj_gemm<float, 0><<<512, 256, 0, stream>>>(value, Wt + 2 * 1048576, bv, (void*)Vb);
  attention_kernel<<<256, 256, 0, stream>>>(Qb, Kt, Vb, Cb, out_attn);
  proj_gemm<_Float16, 2><<<512, 256, 0, stream>>>(Cb, Wt + 3 * 1048576, bo, (void*)prj);
  ln_kernel<<<4096, 256, 0, stream>>>(query, prj, gamma, beta, out_ln);
}
